// RNN_84086869721315
// MI455X (gfx1250) — matrix operand order repaired and checked by reading, not compiled, not run on hardware
//
#include <hip/hip_runtime.h>
#include <hip/hip_bf16.h>

typedef _Float16 h16;
typedef __attribute__((ext_vector_type(16))) _Float16 v16h;
typedef __attribute__((ext_vector_type(8)))  _Float16 v8h;
typedef __attribute__((ext_vector_type(8)))  float    v8f;

#define TT 512
#define BB 32
#define GG 768
#define HH 512
#define POUT 256

static __device__ __forceinline__ v16h mk16(v8h lo, v8h hi){
  return __builtin_shufflevector(lo, hi, 0,1,2,3,4,5,6,7,8,9,10,11,12,13,14,15);
}
static __device__ __forceinline__ v8f wmma_f16(v16h a, v16h b, v8f c){
  return __builtin_amdgcn_wmma_f32_16x16x32_f16(false, a, false, b, (short)0, c, false, false);
}
// A fragment (16x32 f16): X row-major [*, ld], tile at (m0,k0).
// ISA layout: lanes 0-15 row m0+lane hold K 0-7 & 16-23; lanes 16-31 hold K 8-15 & 24-31.
static __device__ __forceinline__ v16h fragA(const h16* X, int ld, int m0, int k0, int lane){
  const h16* p = X + (size_t)(m0 + (lane & 15)) * ld + k0 + ((lane >> 4) << 3);
  v8h lo = *(const v8h*)p;
  v8h hi = *(const v8h*)(p + 16);
  return mk16(lo, hi);
}
// B fragment (32x16 f16): W stored [N, ld] row-major (i.e. W.T columns are W rows).
// ISA layout: lanes 0-15 col n0+lane hold K 0-7 & 16-23; lanes 16-31 hold K 8-15 & 24-31.
static __device__ __forceinline__ v16h fragB(const h16* W, int ld, int n0, int k0, int lane){
  const h16* p = W + (size_t)(n0 + (lane & 15)) * ld + k0 + ((lane >> 4) << 3);
  v8h lo = *(const v8h*)p;
  v8h hi = *(const v8h*)(p + 16);
  return mk16(lo, hi);
}
static __device__ __forceinline__ float sigm(float x){ return 1.f/(1.f+__expf(-x)); }

// ---------------- small utility kernels ----------------

// f32 -> f16 convert of a (possibly strided) sub-matrix
__global__ void cvt_sub(h16* __restrict__ dst, const float* __restrict__ src,
                        int rows, int cols, int sld, int soff){
  size_t n = (size_t)rows * cols;
  for (size_t i = blockIdx.x*(size_t)blockDim.x + threadIdx.x; i < n;
       i += (size_t)gridDim.x*blockDim.x){
    int r = (int)(i / (size_t)cols), c = (int)(i % (size_t)cols);
    dst[i] = (h16)src[(size_t)r*sld + soff + c];
  }
}

// p_tf, teacher-forcing flags, prev0 = initial_pose + 0.01*noise
__global__ void prep(const int* __restrict__ epoch, const int* __restrict__ totep,
                     const float* __restrict__ tfu, const float* __restrict__ noise,
                     const float* __restrict__ init_pose,
                     float* __restrict__ d_out, int* __restrict__ flags,
                     h16* __restrict__ prev0){
  float pg  = (float)epoch[0] / (float)totep[0];
  float ptf = 1.f / (1.f + __expf(10.f*(pg - 0.5f)));
  size_t tid = blockIdx.x*(size_t)blockDim.x + threadIdx.x;
  size_t stride = (size_t)gridDim.x*blockDim.x;
  if (tid == 0) d_out[(size_t)BB*TT*POUT] = ptf;
  for (size_t t = tid; t < TT; t += stride) flags[t] = (tfu[t] < ptf) ? 1 : 0;
  for (size_t i = tid; i < (size_t)BB*POUT; i += stride)
    prev0[i] = (h16)(init_pose[i % POUT] + 0.01f*noise[i]);
}

// ---------------- generic WMMA GEMM: Out[r,n] = sum_k A[r,k]*W[n,k] + bias[n] ----------------
// block = 256 threads (8 waves, 2x4), block tile 32(M) x 256(N).
// amode==1: logical row r = t*BB+b maps to physical row b*TT+t (gloss [B,T,G] transpose view).
__global__ void __launch_bounds__(256) gemm_f16(
    const h16* __restrict__ A, int lda, int amode,
    const h16* __restrict__ W, int K,
    const float* __restrict__ bias,
    h16* __restrict__ Out, int N)
{
  int tid = threadIdx.x, lane = tid & 31, wv = tid >> 5;
  int m0 = blockIdx.x*32 + (wv >> 2)*16;
  int n0 = blockIdx.y*256 + (wv & 3)*64;
  int r  = m0 + (lane & 15);
  int pr = amode ? ((r & (BB-1))*TT + (r >> 5)) : r;
  const h16* arow = A + (size_t)pr*lda + ((lane >> 4) << 3);
  v8f c0 = {}, c1 = {}, c2 = {}, c3 = {};
  for (int k0 = 0; k0 < K; k0 += 32){
    v8h lo = *(const v8h*)(arow + k0);
    v8h hi = *(const v8h*)(arow + k0 + 16);
    v16h a = mk16(lo, hi);
    c0 = wmma_f16(a, fragB(W, K, n0+ 0, k0, lane), c0);
    c1 = wmma_f16(a, fragB(W, K, n0+16, k0, lane), c1);
    c2 = wmma_f16(a, fragB(W, K, n0+32, k0, lane), c2);
    c3 = wmma_f16(a, fragB(W, K, n0+48, k0, lane), c3);
  }
  int cb  = n0 + (lane & 15);
  int hi8 = (lane >> 4) << 3;
  float b0 = bias ? bias[cb]    : 0.f;
  float b1 = bias ? bias[cb+16] : 0.f;
  float b2 = bias ? bias[cb+32] : 0.f;
  float b3 = bias ? bias[cb+48] : 0.f;
  #pragma unroll
  for (int i = 0; i < 8; i++){
    h16* o = Out + (size_t)(m0 + hi8 + i)*N;
    o[cb]    = (h16)(c0[i] + b0);
    o[cb+16] = (h16)(c1[i] + b1);
    o[cb+32] = (h16)(c2[i] + b2);
    o[cb+48] = (h16)(c3[i] + b3);
  }
}

// ---------------- persistent GRU recurrence (one block per direction) ----------------
// gh = h @ Wh.T per step via WMMA. Wave w owns cols {g*512 + [16w,16w+16)}, so its
// accumulators hold exactly r/z/n pre-activations for its h-slice. h master copy in f32 regs.
#define HPAD 520
__global__ void __launch_bounds__(1024) gru_seq(
    const h16* __restrict__ GIf, const h16* __restrict__ GIb,
    const h16* __restrict__ Whf, const h16* __restrict__ Whb,
    const float* __restrict__ bhf_, const float* __restrict__ bhb_,
    h16* __restrict__ Y, h16* __restrict__ ENC)
{
  const int dir = blockIdx.x;
  const h16*  GI = dir ? GIb  : GIf;
  const h16*  Wh = dir ? Whb  : Whf;
  const float* bh = dir ? bhb_ : bhf_;
  __shared__ h16 sH[32*HPAD];
  int tid = threadIdx.x, lane = tid & 31, wv = tid >> 5;
  int j0 = wv*16, jj = j0 + (lane & 15), hi8 = (lane >> 4) << 3;
  float bhr = bh[jj], bhz = bh[512+jj], bhn = bh[1024+jj];
  for (int i = tid; i < 32*HPAD; i += 1024) sH[i] = (h16)0.f;
  float hold[16];
  #pragma unroll
  for (int i = 0; i < 16; i++) hold[i] = 0.f;
  __syncthreads();

  for (int s = 0; s < TT; s++){
    int t = dir ? (TT-1-s) : s;
    // prefetch next step's gi slab (96KB = 1024 threads * 96B) while WMMAs run
    if (s + 1 < TT){
      int tn = dir ? (TT-2-s) : (s+1);
      const h16* gnext = GI + (size_t)tn*BB*1536 + (size_t)tid*48;
      __builtin_prefetch(gnext, 0, 3);
    }
    v8f zero = {};
    v8f acc[2][3] = {{zero,zero,zero},{zero,zero,zero}};
    for (int k0 = 0; k0 < 512; k0 += 32){
      v16h A0 = fragA(sH, HPAD,  0, k0, lane);
      v16h A1 = fragA(sH, HPAD, 16, k0, lane);
      v16h B0 = fragB(Wh, 512,        j0, k0, lane);
      v16h B1 = fragB(Wh, 512,  512 + j0, k0, lane);
      v16h B2 = fragB(Wh, 512, 1024 + j0, k0, lane);
      acc[0][0] = wmma_f16(A0, B0, acc[0][0]);
      acc[1][0] = wmma_f16(A1, B0, acc[1][0]);
      acc[0][1] = wmma_f16(A0, B1, acc[0][1]);
      acc[1][1] = wmma_f16(A1, B1, acc[1][1]);
      acc[0][2] = wmma_f16(A0, B2, acc[0][2]);
      acc[1][2] = wmma_f16(A1, B2, acc[1][2]);
    }
    __syncthreads();   // all waves done reading old h
    const h16* git = GI + (size_t)t*BB*1536;
    #pragma unroll
    for (int mt = 0; mt < 2; mt++)
      #pragma unroll
      for (int i = 0; i < 8; i++){
        int b = mt*16 + hi8 + i;
        const h16* gb = git + (size_t)b*1536;
        float r  = sigm ((float)gb[jj]        + acc[mt][0][i] + bhr);
        float z  = sigm ((float)gb[512+jj]    + acc[mt][1][i] + bhz);
        float n  = tanhf((float)gb[1024+jj]   + r*(acc[mt][2][i] + bhn));
        float hn = (1.f - z)*n + z*hold[mt*8+i];
        hold[mt*8+i] = hn;
        sH[b*HPAD + jj] = (h16)hn;
        Y[((size_t)t*BB + b)*1024 + dir*512 + jj] = (h16)hn;
      }
    __syncthreads();   // new h visible for next step
  }
  if (ENC){
    #pragma unroll
    for (int mt = 0; mt < 2; mt++)
      #pragma unroll
      for (int i = 0; i < 8; i++){
        int b = mt*16 + hi8 + i;
        ENC[(size_t)b*1024 + dir*512 + jj] = (h16)hold[mt*8+i];
      }
  }
}

// ---------------- fused decoder: GRU + (folded) attention + LN + FC, 512 steps ----------------
__global__ void __launch_bounds__(1024) decoder_seq(
    const h16* __restrict__ GID,    // [T,B,1536] text-side gi + bi_dec
    const h16* __restrict__ CTXP,   // [T,B,512]  ctx@W_ap2.T + b_ap (folded attention)
    const h16* __restrict__ H0,     // [32,512]
    const h16* __restrict__ PRV0,   // [32,256]
    const h16* __restrict__ Whd,    // [1536,512]
    const h16* __restrict__ Wip,    // [1536,256] (Wi_dec cols 0..255)
    const h16* __restrict__ Wap1,   // [512,512]  (W_ap cols 0..511)
    const h16* __restrict__ Wfc,    // [256,512]
    const float* __restrict__ bhd,
    const float* __restrict__ lng, const float* __restrict__ lnb,
    const float* __restrict__ bfc,
    const float* __restrict__ gt,   // [B,T,P]
    const int* __restrict__ flags,  // [T]
    float* __restrict__ out)        // [B,T,P]
{
  __shared__ h16 sH[32*HPAD];
  __shared__ h16 sPrev[32*264];
  __shared__ h16 sF[32*HPAD];
  int tid = threadIdx.x, lane = tid & 31, wv = tid >> 5;
  int j0 = wv*16, jj = j0 + (lane & 15), hi8 = (lane >> 4) << 3;
  float bhr = bhd[jj], bhz = bhd[512+jj], bhn = bhd[1024+jj];
  // FC tile mapping: wave -> (m-tile, n-tile)
  int fmt = wv & 1, fnt = wv >> 1;
  int fcol = fnt*16 + (lane & 15);
  float bfc_c = bfc[fcol];
  // LN hoists: wave wv owns row wv; lane owns cols lane+32q
  float lg[16], lb[16];
  #pragma unroll
  for (int q = 0; q < 16; q++){ lg[q] = lng[lane + 32*q]; lb[q] = lnb[lane + 32*q]; }
  for (int i = tid; i < 32*512; i += 1024){ sH[(i>>9)*HPAD + (i&511)] = H0[i]; }
  for (int i = tid; i < 32*256; i += 1024){ sPrev[(i>>8)*264 + (i&255)] = PRV0[i]; }
  float hold[16];
  #pragma unroll
  for (int mt = 0; mt < 2; mt++)
    #pragma unroll
    for (int i = 0; i < 8; i++)
      hold[mt*8+i] = (float)H0[(size_t)(mt*16 + hi8 + i)*512 + jj];
  __syncthreads();

  for (int t = 0; t < TT; t++){
    bool tf = flags[t] != 0;
    // prefetch step t+1 operand slabs while this step's WMMAs run:
    //  GID slab 96KB (1024*96B), CTXP slab 32KB (1024*32B), gt rows 32x1KB (256 lines)
    if (t + 1 < TT){
      __builtin_prefetch(GID  + (size_t)(t+1)*BB*1536 + (size_t)tid*48, 0, 3);
      __builtin_prefetch(CTXP + (size_t)(t+1)*BB*512  + (size_t)tid*16, 0, 3);
      if (tid < 256){
        int pb = tid >> 3, seg = tid & 7;
        __builtin_prefetch(gt + ((size_t)pb*TT + (t+1))*POUT + seg*32, 0, 3);
      }
    }
    v8f zero = {};
    // ---- gate GEMMs: r/z accumulate prev- and h- contributions jointly;
    //      n-gate keeps gi part (prev) and gh part (h) separate.
    v8f aR[2]={zero,zero}, aZ[2]={zero,zero}, aNi[2]={zero,zero}, aNh[2]={zero,zero};
    for (int k0 = 0; k0 < 512; k0 += 32){
      v16h A0 = fragA(sH, HPAD,  0, k0, lane);
      v16h A1 = fragA(sH, HPAD, 16, k0, lane);
      v16h Br = fragB(Whd, 512,        j0, k0, lane);
      v16h Bz = fragB(Whd, 512,  512 + j0, k0, lane);
      v16h Bn = fragB(Whd, 512, 1024 + j0, k0, lane);
      aR[0]=wmma_f16(A0,Br,aR[0]); aR[1]=wmma_f16(A1,Br,aR[1]);
      aZ[0]=wmma_f16(A0,Bz,aZ[0]); aZ[1]=wmma_f16(A1,Bz,aZ[1]);
      aNh[0]=wmma_f16(A0,Bn,aNh[0]); aNh[1]=wmma_f16(A1,Bn,aNh[1]);
    }
    for (int k0 = 0; k0 < 256; k0 += 32){
      v16h A0 = fragA(sPrev, 264,  0, k0, lane);
      v16h A1 = fragA(sPrev, 264, 16, k0, lane);
      v16h Br = fragB(Wip, 256,        j0, k0, lane);
      v16h Bz = fragB(Wip, 256,  512 + j0, k0, lane);
      v16h Bn = fragB(Wip, 256, 1024 + j0, k0, lane);
      aR[0]=wmma_f16(A0,Br,aR[0]); aR[1]=wmma_f16(A1,Br,aR[1]);
      aZ[0]=wmma_f16(A0,Bz,aZ[0]); aZ[1]=wmma_f16(A1,Bz,aZ[1]);
      aNi[0]=wmma_f16(A0,Bn,aNi[0]); aNi[1]=wmma_f16(A1,Bn,aNi[1]);
    }
    __syncthreads();
    // ---- gates, update h (out == new h)
    const h16* git = GID + (size_t)t*BB*1536;
    #pragma unroll
    for (int mt = 0; mt < 2; mt++)
      #pragma unroll
      for (int i = 0; i < 8; i++){
        int b = mt*16 + hi8 + i;
        const h16* gb = git + (size_t)b*1536;
        float r  = sigm ((float)gb[jj]      + aR[mt][i] + bhr);
        float z  = sigm ((float)gb[512+jj]  + aZ[mt][i] + bhz);
        float n  = tanhf((float)gb[1024+jj] + aNi[mt][i] + r*(aNh[mt][i] + bhn));
        float hn = (1.f - z)*n + z*hold[mt*8+i];
        hold[mt*8+i] = hn;
        sH[b*HPAD + jj] = (h16)hn;
      }
    __syncthreads();
    // ---- projected = out@W_ap1.T + CTXP[t] (+b_ap folded) ; final = projected + out
    v8f ap[2] = {zero, zero};
    for (int k0 = 0; k0 < 512; k0 += 32){
      v16h A0 = fragA(sH, HPAD,  0, k0, lane);
      v16h A1 = fragA(sH, HPAD, 16, k0, lane);
      v16h Bp = fragB(Wap1, 512, j0, k0, lane);
      ap[0] = wmma_f16(A0, Bp, ap[0]);
      ap[1] = wmma_f16(A1, Bp, ap[1]);
    }
    const h16* ctxt = CTXP + (size_t)t*BB*512;
    #pragma unroll
    for (int mt = 0; mt < 2; mt++)
      #pragma unroll
      for (int i = 0; i < 8; i++){
        int b = mt*16 + hi8 + i;
        float fin = ap[mt][i] + (float)ctxt[(size_t)b*512 + jj] + hold[mt*8+i];
        sF[b*HPAD + jj] = (h16)fin;
      }
    __syncthreads();
    // ---- LayerNorm over H (wave wv <-> row wv), normed written back to sF
    {
      int b = wv;
      float vals[16]; float s = 0.f, ss = 0.f;
      #pragma unroll
      for (int q = 0; q < 16; q++){
        float v = (float)sF[b*HPAD + lane + 32*q];
        vals[q] = v; s += v; ss += v*v;
      }
      #pragma unroll
      for (int m = 16; m >= 1; m >>= 1){ s += __shfl_xor(s, m); ss += __shfl_xor(ss, m); }
      float mu  = s * (1.f/512.f);
      float var = ss * (1.f/512.f) - mu*mu;
      float inv = rsqrtf(var + 1e-5f);
      #pragma unroll
      for (int q = 0; q < 16; q++)
        sF[b*HPAD + lane + 32*q] = (h16)((vals[q]-mu)*inv*lg[q] + lb[q]);
    }
    __syncthreads();
    // ---- pose = normed@W_fc.T + b_fc ; teacher forcing for next prev
    v8f af = zero;
    for (int k0 = 0; k0 < 512; k0 += 32){
      v16h A0 = fragA(sF, HPAD, fmt*16, k0, lane);
      v16h Bf = fragB(Wfc, 512, fnt*16, k0, lane);
      af = wmma_f16(A0, Bf, af);
    }
    #pragma unroll
    for (int i = 0; i < 8; i++){
      int b = fmt*16 + hi8 + i;
      float pose = af[i] + bfc_c;
      size_t oidx = ((size_t)b*TT + t)*POUT + fcol;
      out[oidx] = pose;
      float pv = tf ? gt[oidx] : pose;
      sPrev[b*264 + fcol] = (h16)pv;
    }
    __syncthreads();
  }
}

// ---------------- host orchestration ----------------
extern "C" void kernel_launch(void* const* d_in, const int* in_sizes, int n_in,
                              void* d_out, int out_size, void* d_ws, size_t ws_size,
                              hipStream_t stream){
  (void)in_sizes; (void)n_in; (void)out_size; (void)ws_size;
  const float* gloss   = (const float*)d_in[0];
  const float* gtruth  = (const float*)d_in[1];
  const int*   epoch   = (const int*)d_in[2];
  const int*   totep   = (const int*)d_in[3];
  const float* noise   = (const float*)d_in[4];
  const float* tfu     = (const float*)d_in[5];
  const float* Wi_l0f  = (const float*)d_in[6];
  const float* Wh_l0f  = (const float*)d_in[7];
  const float* bi_l0f  = (const float*)d_in[8];
  const float* bh_l0f  = (const float*)d_in[9];
  const float* Wi_l0b  = (const float*)d_in[10];
  const float* Wh_l0b  = (const float*)d_in[11];
  const float* bi_l0b  = (const float*)d_in[12];
  const float* bh_l0b  = (const float*)d_in[13];
  const float* Wi_l1f  = (const float*)d_in[14];
  const float* Wh_l1f  = (const float*)d_in[15];
  const float* bi_l1f  = (const float*)d_in[16];
  const float* bh_l1f  = (const float*)d_in[17];
  const float* Wi_l1b  = (const float*)d_in[18];
  const float* Wh_l1b  = (const float*)d_in[19];
  const float* bi_l1b  = (const float*)d_in[20];
  const float* bh_l1b  = (const float*)d_in[21];
  const float* W_tp    = (const float*)d_in[22];
  const float* b_tp    = (const float*)d_in[23];
  const float* W_e2d   = (const float*)d_in[24];
  const float* b_e2d   = (const float*)d_in[25];
  const float* Wi_dec  = (const float*)d_in[26];
  const float* Wh_dec  = (const float*)d_in[27];
  const float* bi_dec  = (const float*)d_in[28];
  const float* bh_dec  = (const float*)d_in[29];
  // d_in[30..33] = Wq,bq,Wk,bk : mathematically dead (softmax over 1 key == 1)
  const float* Wv      = (const float*)d_in[34];
  const float* bv      = (const float*)d_in[35];
  const float* Wo      = (const float*)d_in[36];
  const float* bo      = (const float*)d_in[37];
  const float* W_ap    = (const float*)d_in[38];
  const float* b_ap    = (const float*)d_in[39];
  const float* ln_g    = (const float*)d_in[40];
  const float* ln_b    = (const float*)d_in[41];
  const float* W_fc    = (const float*)d_in[42];
  const float* b_fc    = (const float*)d_in[43];
  const float* init_ps = (const float*)d_in[44];
  float* out = (float*)d_out;

  // bump allocator over d_ws (total ~254 MB)
  size_t off = 0;
  auto alloc = [&](size_t elems)->h16*{
    h16* p = (h16*)((char*)d_ws + off);
    off += ((elems*sizeof(h16)) + 255) & ~(size_t)255;
    return p;
  };
  h16* X0h   = alloc((size_t)BB*TT*GG);
  h16* wWi0f = alloc(1536*768);  h16* wWh0f = alloc(1536*512);
  h16* wWi0b = alloc(1536*768);  h16* wWh0b = alloc(1536*512);
  h16* wWi1f = alloc(1536*1024); h16* wWh1f = alloc(1536*512);
  h16* wWi1b = alloc(1536*1024); h16* wWh1b = alloc(1536*512);
  h16* wWtp  = alloc(512*1024);
  h16* wWe2d = alloc(512*1024);
  h16* wWip  = alloc(1536*256);
  h16* wWix  = alloc(1536*1024);
  h16* wWhd  = alloc(1536*512);
  h16* wWv   = alloc(512*512);
  h16* wWo   = alloc(512*512);
  h16* wWap1 = alloc(512*512);
  h16* wWap2 = alloc(512*512);
  h16* wWfc  = alloc(256*512);
  h16* GIF   = alloc((size_t)TT*BB*1536);   // reused: L0f gi -> L1f gi -> decoder text gi
  h16* GIB   = alloc((size_t)TT*BB*1536);   // reused: L0b gi -> L1b gi
  h16* Y0    = alloc((size_t)TT*BB*1024);
  h16* TEXT  = alloc((size_t)TT*BB*1024);
  h16* PPA   = alloc((size_t)TT*BB*512);    // ping
  h16* PPB   = alloc((size_t)TT*BB*512);    // pong (final CTXP lives here)
  h16* ENC   = alloc(32*1024);
  h16* H0b   = alloc(32*512);
  h16* PRV0  = alloc(32*256);
  int* flags = (int*)((char*)d_ws + off);
  off += (512*sizeof(int) + 255) & ~(size_t)255;

  prep<<<32, 256, 0, stream>>>(epoch, totep, tfu, noise, init_ps, out, flags, PRV0);

  auto cvt = [&](h16* dst, const float* src, int rows, int cols, int sld, int soff){
    size_t n = (size_t)rows*cols;
    int blocks = (int)((n + 255)/256); if (blocks > 16384) blocks = 16384;
    cvt_sub<<<blocks, 256, 0, stream>>>(dst, src, rows, cols, sld, soff);
  };
  cvt(X0h,   gloss,  BB*TT, GG, GG, 0);
  cvt(wWi0f, Wi_l0f, 1536, 768,  768,  0);
  cvt(wWh0f, Wh_l0f, 1536, 512,  512,  0);
  cvt(wWi0b, Wi_l0b, 1536, 768,  768,  0);
  cvt(wWh0b, Wh_l0b, 1536, 512,  512,  0);
  cvt(wWi1f, Wi_l1f, 1536, 1024, 1024, 0);
  cvt(wWh1f, Wh_l1f, 1536, 512,  512,  0);
  cvt(wWi1b, Wi_l1b, 1536, 1024, 1024, 0);
  cvt(wWh1b, Wh_l1b, 1536, 512,  512,  0);
  cvt(wWtp,  W_tp,   512,  1024, 1024, 0);
  cvt(wWe2d, W_e2d,  512,  1024, 1024, 0);
  cvt(wWip,  Wi_dec, 1536, 256,  1280, 0);    // prev columns
  cvt(wWix,  Wi_dec, 1536, 1024, 1280, 256);  // text columns
  cvt(wWhd,  Wh_dec, 1536, 512,  512,  0);
  cvt(wWv,   Wv,     512,  512,  512,  0);
  cvt(wWo,   Wo,     512,  512,  512,  0);
  cvt(wWap1, W_ap,   512,  512,  1024, 0);    // "out" columns
  cvt(wWap2, W_ap,   512,  512,  1024, 512);  // "ctx" columns
  cvt(wWfc,  W_fc,   256,  512,  512,  0);

  dim3 blk(256);
  // encoder layer 0: input projections (amode=1 handles [B,T,G] -> [T,B,G] view)
  gemm_f16<<<dim3(512,6), blk, 0, stream>>>(X0h, GG, 1, wWi0f, GG, bi_l0f, GIF, 1536);
  gemm_f16<<<dim3(512,6), blk, 0, stream>>>(X0h, GG, 1, wWi0b, GG, bi_l0b, GIB, 1536);
  gru_seq<<<2, 1024, 0, stream>>>(GIF, GIB, wWh0f, wWh0b, bh_l0f, bh_l0b, Y0, (h16*)nullptr);
  // encoder layer 1
  gemm_f16<<<dim3(512,6), blk, 0, stream>>>(Y0, 1024, 0, wWi1f, 1024, bi_l1f, GIF, 1536);
  gemm_f16<<<dim3(512,6), blk, 0, stream>>>(Y0, 1024, 0, wWi1b, 1024, bi_l1b, GIB, 1536);
  gru_seq<<<2, 1024, 0, stream>>>(GIF, GIB, wWh1f, wWh1b, bh_l1f, bh_l1b, TEXT, ENC);
  // decoder precomputes (all batched over T*B rows)
  gemm_f16<<<dim3(512,6), blk, 0, stream>>>(TEXT, 1024, 0, wWix,  1024, bi_dec, GIF, 1536);
  gemm_f16<<<dim3(512,2), blk, 0, stream>>>(TEXT, 1024, 0, wWtp,  1024, b_tp,   PPA, 512);
  gemm_f16<<<dim3(512,2), blk, 0, stream>>>(PPA,  512,  0, wWv,   512,  bv,     PPB, 512);
  gemm_f16<<<dim3(512,2), blk, 0, stream>>>(PPB,  512,  0, wWo,   512,  bo,     PPA, 512);
  gemm_f16<<<dim3(512,2), blk, 0, stream>>>(PPA,  512,  0, wWap2, 512,  b_ap,   PPB, 512);
  gemm_f16<<<dim3(1,2),   blk, 0, stream>>>(ENC,  1024, 0, wWe2d, 1024, b_e2d,  H0b, 512);
  // fused decoder scan
  decoder_seq<<<1, 1024, 0, stream>>>(GIF, PPB, H0b, PRV0, wWhd, wWip, wWap1, wWfc,
                                      bh_dec, ln_g, ln_b, b_fc, gtruth, flags, out);
}
